// BasicGGNN_3736621548149
// MI455X (gfx1250) — matrix operand order repaired and checked by reading, not compiled, not run on hardware
//
#include <hip/hip_runtime.h>
#include <hip/hip_bf16.h>

// ---------------- constants ----------------
#define NN      15000
#define EE      240000
#define IN_DIM  100
#define HID     128
#define NTYPE   13
#define NSTEPS  6
#define NGRAPH  64
#define NCLS    10
#define MP      15008          // N padded to multiple of 16
#define MTILES  (MP / 16)      // 938

typedef __attribute__((ext_vector_type(16))) __bf16 bf16x16;
typedef __attribute__((ext_vector_type(8)))  __bf16 bf16x8;
typedef __attribute__((ext_vector_type(4)))  __bf16 bf16x4;
typedef __attribute__((ext_vector_type(8)))  float  f32x8;
typedef __attribute__((ext_vector_type(4)))  float  f32x4;

// ---------------- native f32 atomic adds (CDNA5 global_atomic_add_f32) ------
// Single asm block, one base address, immediate offsets: avoids both the
// CAS-loop expansion of atomicAdd(float*) and per-op address recomputation.
__device__ __forceinline__ void global_fadd_x4(float* p, float v0, float v1,
                                               float v2, float v3) {
  asm volatile("global_atomic_add_f32 %0, %1, off\n\t"
               "global_atomic_add_f32 %0, %2, off offset:4\n\t"
               "global_atomic_add_f32 %0, %3, off offset:8\n\t"
               "global_atomic_add_f32 %0, %4, off offset:12"
               :
               : "v"(p), "v"(v0), "v"(v1), "v"(v2), "v"(v3)
               : "memory");
}

// ---------------- WMMA helpers ----------------
// A fragment: 16x32 bf16, row m = lane&15, K chunks [kb, kb+8) and [kb+16, kb+24)
// where kb = k0 + (lane>=16 ? 8 : 0).  Two contiguous 16-byte loads per lane.
__device__ __forceinline__ bf16x16 load_a_frag(const __bf16* __restrict__ A,
                                               int lda, int m0, int k0, int lane) {
  int m  = m0 + (lane & 15);
  int kb = k0 + ((lane >> 4) << 3);
  const __bf16* p = A + (size_t)m * lda + kb;
  bf16x8 lo = *reinterpret_cast<const bf16x8*>(p);
  bf16x8 hi = *reinterpret_cast<const bf16x8*>(p + 16);
  bf16x16 r;
#pragma unroll
  for (int i = 0; i < 8; ++i) { r[i] = lo[i]; r[i + 8] = hi[i]; }
  return r;
}

// Same A fragment but sourced from an f32 buffer, converting to bf16 in
// registers (v_cvt_pk_bf16_f32) -- avoids a separate conversion pass.
__device__ __forceinline__ bf16x16 load_a_frag_f32(const float* __restrict__ A,
                                                   int lda, int m0, int k0, int lane) {
  int m  = m0 + (lane & 15);
  int kb = k0 + ((lane >> 4) << 3);
  const float* p = A + (size_t)m * lda + kb;
  f32x8 lo = *reinterpret_cast<const f32x8*>(p);
  f32x8 hi = *reinterpret_cast<const f32x8*>(p + 16);
  bf16x16 r;
#pragma unroll
  for (int i = 0; i < 8; ++i) { r[i] = (__bf16)lo[i]; r[i + 8] = (__bf16)hi[i]; }
  return r;
}

// B fragment: 32x16 bf16 from PRE-TRANSPOSED weights BT[n][k] (row-major N x K).
// column n = lane&15, K chunks [kb, kb+8) and [kb+16, kb+24), kb = k0 + (lane>=16 ? 8 : 0).
__device__ __forceinline__ bf16x16 load_b_frag(const __bf16* __restrict__ BT,
                                               int ldb, int n0, int k0, int lane) {
  int n  = n0 + (lane & 15);
  int kb = k0 + ((lane >> 4) << 3);
  const __bf16* p = BT + (size_t)n * ldb + kb;
  bf16x8 lo = *reinterpret_cast<const bf16x8*>(p);
  bf16x8 hi = *reinterpret_cast<const bf16x8*>(p + 16);
  bf16x16 r;
#pragma unroll
  for (int i = 0; i < 8; ++i) { r[i] = lo[i]; r[i + 8] = hi[i]; }
  return r;
}

__device__ __forceinline__ f32x8 wmma_bf16(bf16x16 a, bf16x16 b, f32x8 c) {
  return __builtin_amdgcn_wmma_f32_16x16x32_bf16(false, a, false, b,
                                                 (short)0, c, false, false);
}

// ---------------- kernels ----------------

// Convert weights to bf16; W_e transposed to WeT[t][n][k] = W_e[t][k][n].
__global__ void ggnn_prep_weights(const float* __restrict__ We,
                                  const float* __restrict__ Wih,
                                  const float* __restrict__ Whh,
                                  __bf16* __restrict__ WeT,
                                  __bf16* __restrict__ Wihb,
                                  __bf16* __restrict__ Whhb) {
  int i = blockIdx.x * blockDim.x + threadIdx.x;
  if (i < NTYPE * HID * HID) {
    int t = i / (HID * HID);
    int r = i - t * HID * HID;
    int n = r >> 7;          // output col
    int k = r & 127;         // reduction dim
    WeT[i] = (__bf16)We[(size_t)t * HID * HID + (size_t)k * HID + n];
  }
  if (i < 3 * HID * HID) {   // gi = a @ W_ih^T  => BT[n][k] = W_ih[n][k] (copy)
    Wihb[i] = (__bf16)Wih[i];
    Whhb[i] = (__bf16)Whh[i];
  }
}

// h = zero-padded feat (both f32 and bf16 mirrors), padded rows zeroed.
__global__ void ggnn_init_h(const float* __restrict__ feat,
                            float* __restrict__ h, __bf16* __restrict__ hb) {
  int i = blockIdx.x * blockDim.x + threadIdx.x;
  if (i >= MP * HID) return;
  int m = i >> 7, d = i & 127;
  float v = (m < NN && d < IN_DIM) ? feat[(size_t)m * IN_DIM + d] : 0.0f;
  h[i]  = v;
  hb[i] = (__bf16)v;
}

__global__ void ggnn_zero_f32(float* __restrict__ x, int n) {
  int i = blockIdx.x * blockDim.x + threadIdx.x;
  if (i < n) x[i] = 0.0f;
}

// Wh[t][m][n] = (h @ W_e[t])[m][n] + b_e[t][n], bf16 output.
// grid = (MTILES, NTYPE), block = 256 (8 waves); wave w does columns [16w,16w+16).
__global__ void ggnn_etype_gemm(const __bf16* __restrict__ hb,
                                const __bf16* __restrict__ WeT,
                                const float* __restrict__ be,
                                __bf16* __restrict__ Wh) {
  int m0   = blockIdx.x * 16;
  int t    = blockIdx.y;
  int wave = threadIdx.x >> 5;
  int lane = threadIdx.x & 31;
  int n0   = wave * 16;
  const __bf16* BT = WeT + (size_t)t * HID * HID;
  f32x8 acc = {};
#pragma unroll
  for (int kc = 0; kc < 4; ++kc) {
    bf16x16 a = load_a_frag(hb, HID, m0, kc * 32, lane);
    bf16x16 b = load_b_frag(BT, HID, n0, kc * 32, lane);
    acc = wmma_bf16(a, b, acc);
  }
  int n = n0 + (lane & 15);
  float bias = be[t * HID + n];
  __bf16* out = Wh + ((size_t)t * MP + m0) * HID;
#pragma unroll
  for (int v = 0; v < 8; ++v) {
    int m = ((lane >> 4) << 3) + v;
    out[(size_t)m * HID + n] = (__bf16)(acc[v] + bias);
  }
}

// One wave per edge: gather bf16 message row, scatter native f32 atomics
// (4 per lane, one base address, immediate offsets).
__global__ void ggnn_edge_scatter(const int* __restrict__ src,
                                  const int* __restrict__ dst,
                                  const int* __restrict__ ety,
                                  const __bf16* __restrict__ Wh,
                                  float* __restrict__ a) {
  int e = (blockIdx.x * blockDim.x + threadIdx.x) >> 5;
  if (e >= EE) return;
  int lane = threadIdx.x & 31;
  int t = ety[e], s = src[e], d = dst[e];
  const __bf16* msg = Wh + (((size_t)t * MP + s) * HID) + lane * 4;
  float* ap = a + (size_t)d * HID + lane * 4;
  bf16x4 v = *reinterpret_cast<const bf16x4*>(msg);
  global_fadd_x4(ap, (float)v[0], (float)v[1], (float)v[2], (float)v[3]);
}

// Fused GRU: per 16-row block, 8 waves compute all 48 tiles of gi/gh (WMMA),
// stage in LDS, then fused gating updates h (f32 + bf16) in place.
// A for gi is read from the f32 accumulator with in-register bf16 convert.
__global__ void ggnn_gru_fused(const float* __restrict__ a,
                               __bf16* __restrict__ hb,
                               const __bf16* __restrict__ Wihb,
                               const __bf16* __restrict__ Whhb,
                               const float* __restrict__ bih,
                               const float* __restrict__ bhh,
                               float* __restrict__ h) {
  __shared__ float gi[16 * 384];
  __shared__ float gh[16 * 384];
  int m0   = blockIdx.x * 16;
  int wave = threadIdx.x >> 5;
  int lane = threadIdx.x & 31;

#pragma unroll
  for (int i = 0; i < 6; ++i) {
    int tt   = wave * 6 + i;       // 0..47
    bool isH = tt >= 24;
    int nt   = isH ? tt - 24 : tt; // 0..23 -> column tile in [0,384)
    const __bf16* BT = isH ? Whhb : Wihb;
    f32x8 acc = {};
#pragma unroll
    for (int kc = 0; kc < 4; ++kc) {
      bf16x16 av = isH ? load_a_frag(hb, HID, m0, kc * 32, lane)
                       : load_a_frag_f32(a, HID, m0, kc * 32, lane);
      bf16x16 bv = load_b_frag(BT, HID, nt * 16, kc * 32, lane);
      acc = wmma_bf16(av, bv, acc);
    }
    float* buf = isH ? gh : gi;
    int n = nt * 16 + (lane & 15);
#pragma unroll
    for (int v = 0; v < 8; ++v) {
      int m = ((lane >> 4) << 3) + v;
      buf[m * 384 + n] = acc[v];
    }
  }
  __syncthreads();

  for (int idx = threadIdx.x; idx < 16 * HID; idx += blockDim.x) {
    int m = idx >> 7, j = idx & 127;
    float ir = gi[m * 384 + j]        + bih[j];
    float iz = gi[m * 384 + 128 + j]  + bih[128 + j];
    float in_= gi[m * 384 + 256 + j]  + bih[256 + j];
    float hr = gh[m * 384 + j]        + bhh[j];
    float hz = gh[m * 384 + 128 + j]  + bhh[128 + j];
    float hn = gh[m * 384 + 256 + j]  + bhh[256 + j];
    float r  = 1.0f / (1.0f + __expf(-(ir + hr)));
    float z  = 1.0f / (1.0f + __expf(-(iz + hz)));
    float nn = tanhf(in_ + r * hn);
    size_t o = (size_t)(m0 + m) * HID + j;
    float hnew = (1.0f - z) * nn + z * h[o];
    h[o]  = hnew;
    hb[o] = (__bf16)hnew;
  }
}

// Sum node states per graph: each thread handles 4 contiguous columns of one
// node row -> one float4 load, one gid load, one 4-atomic clause.
__global__ void ggnn_readout_acc(const float* __restrict__ h,
                                 const int* __restrict__ gid,
                                 float* __restrict__ hg) {
  int i = blockIdx.x * blockDim.x + threadIdx.x;   // over NN * 32
  if (i >= NN * (HID / 4)) return;
  int m  = i >> 5;
  int j4 = (i & 31) * 4;
  f32x4 v = *reinterpret_cast<const f32x4*>(h + (size_t)m * HID + j4);
  float* p = hg + (size_t)gid[m] * HID + j4;
  global_fadd_x4(p, v[0], v[1], v[2], v[3]);
}

__global__ void ggnn_classify(const float* __restrict__ hg,
                              const float* __restrict__ Wcls,
                              const float* __restrict__ bcls,
                              float* __restrict__ out) {
  int i = blockIdx.x * blockDim.x + threadIdx.x;
  if (i >= NGRAPH * NCLS) return;
  int g = i / NCLS, c = i - g * NCLS;
  float s = bcls[c];
#pragma unroll 4
  for (int d = 0; d < HID; ++d) s += hg[g * HID + d] * Wcls[c * HID + d];
  out[i] = s;
}

// ---------------- host launch ----------------
extern "C" void kernel_launch(void* const* d_in, const int* in_sizes, int n_in,
                              void* d_out, int out_size, void* d_ws, size_t ws_size,
                              hipStream_t stream) {
  (void)in_sizes; (void)n_in; (void)out_size; (void)ws_size;
  const float* feat = (const float*)d_in[0];
  const int*   src  = (const int*)  d_in[1];
  const int*   dst  = (const int*)  d_in[2];
  const int*   ety  = (const int*)  d_in[3];
  const int*   gid  = (const int*)  d_in[4];
  const float* We   = (const float*)d_in[5];
  const float* be   = (const float*)d_in[6];
  const float* Wih  = (const float*)d_in[7];
  const float* Whh  = (const float*)d_in[8];
  const float* bih  = (const float*)d_in[9];
  const float* bhh  = (const float*)d_in[10];
  const float* Wcls = (const float*)d_in[11];
  const float* bcls = (const float*)d_in[12];
  float* out = (float*)d_out;

  // workspace carve (256B aligned)
  char* w = (char*)d_ws;
  auto take = [&](size_t bytes) { void* p = (void*)w; w += (bytes + 255) & ~(size_t)255; return p; };
  float*  h    = (float*) take((size_t)MP * HID * 4);
  __bf16* hb   = (__bf16*)take((size_t)MP * HID * 2);
  float*  a    = (float*) take((size_t)MP * HID * 4);
  __bf16* Wh   = (__bf16*)take((size_t)NTYPE * MP * HID * 2);
  __bf16* WeT  = (__bf16*)take((size_t)NTYPE * HID * HID * 2);
  __bf16* Wihb = (__bf16*)take((size_t)3 * HID * HID * 2);
  __bf16* Whhb = (__bf16*)take((size_t)3 * HID * HID * 2);
  float*  hg   = (float*) take((size_t)NGRAPH * HID * 4);

  const int TB = 256;
  const int nElem   = MP * HID;
  const int gElem   = (nElem + TB - 1) / TB;
  const int gPrep   = (NTYPE * HID * HID + TB - 1) / TB;
  const int gEdge   = (EE * 32 + TB - 1) / TB;
  const int gNode4  = (NN * (HID / 4) + TB - 1) / TB;
  const int gHg     = (NGRAPH * HID + TB - 1) / TB;
  const int gCls    = (NGRAPH * NCLS + TB - 1) / TB;

  ggnn_prep_weights<<<gPrep, TB, 0, stream>>>(We, Wih, Whh, WeT, Wihb, Whhb);
  ggnn_init_h<<<gElem, TB, 0, stream>>>(feat, h, hb);

  for (int step = 0; step < NSTEPS; ++step) {
    ggnn_etype_gemm<<<dim3(MTILES, NTYPE), TB, 0, stream>>>(hb, WeT, be, Wh);
    ggnn_zero_f32<<<gElem, TB, 0, stream>>>(a, nElem);
    ggnn_edge_scatter<<<gEdge, TB, 0, stream>>>(src, dst, ety, Wh, a);
    ggnn_gru_fused<<<MTILES, TB, 0, stream>>>(a, hb, Wihb, Whhb, bih, bhh, h);
  }

  ggnn_zero_f32<<<gHg, TB, 0, stream>>>(hg, NGRAPH * HID);
  ggnn_readout_acc<<<gNode4, TB, 0, stream>>>(h, gid, hg);
  ggnn_classify<<<gCls, TB, 0, stream>>>(hg, Wcls, bcls, out);
}
